// RNN_G_57294863728656
// MI455X (gfx1250) — matrix operand order repaired and checked by reading, not compiled, not run on hardware
//
#include <hip/hip_runtime.h>

// ---------------- problem constants (from reference) ----------------
constexpr int Bn    = 4096;
constexpr int Tn    = 400;
constexpr int Cn    = 42;
constexpr int Hn    = 16;
constexpr int NCn   = 2;
constexpr int TILES = Bn / 16;   // 256 batch tiles (WMMA N = 16)
constexpr int FRAG  = 768;       // packed bf16 elements per (t, tile): 512 + 256
constexpr int TT    = 32;        // timesteps handled per pack block

typedef __attribute__((ext_vector_type(16))) __bf16       v16bf;
typedef __attribute__((ext_vector_type(8)))  float        v8f;
typedef __attribute__((ext_vector_type(8)))  unsigned int v8u;

__device__ __forceinline__ float tanh_f32(float x){
#if __has_builtin(__builtin_amdgcn_tanhf)
  return __builtin_amdgcn_tanhf(x);       // v_tanh_f32 (gfx1250 TRANS op)
#else
  float r;
  asm volatile("v_tanh_f32 %0, %1\n\tv_nop\n\tv_nop" : "=v"(r) : "v"(x));
  return r;
#endif
}

__device__ __forceinline__ v8f wmma_bf16(v16bf a, v16bf b, v8f c){
  // D(f32 16x16) = A(16x32 bf16) * B(32x16 bf16) + C
  return __builtin_amdgcn_wmma_f32_16x16x32_bf16(false, a, false, b,
                                                 (short)0, c, false, false);
}

// AND-mask all 32 bytes of a B fragment with a per-lane mask (branch-free
// replacement for "upper half-wave is K-padding => zero").
__device__ __forceinline__ v16bf mask_frag(v16bf v, unsigned m){
  v8u u = __builtin_bit_cast(v8u, v);
#pragma unroll
  for (int k = 0; k < 8; ++k) u[k] &= m;
  return __builtin_bit_cast(v16bf, u);
}

// D accumulator (transposed state: M=hidden, N=batch; element r at lane L is
// (M=r+8*(L>>4), N=L&15)) -> B operand for the next 16x16x32 WMMA.
// B layout used here (contiguous k, same order as A): lane L holds column
// N=L&15; element j has K=(L>>4)*16+j. Only K<16 is live (H=16): lanes 0..15
// carry all data (rows 8..15 fetched from lane L+16), lanes 16..31 masked to 0.
__device__ __forceinline__ v16bf d_to_bfrag(const float t8[8], int bpidx,
                                            unsigned m){
  float tp[8];
#pragma unroll
  for (int r = 0; r < 8; ++r){
    int v = __builtin_amdgcn_ds_bpermute(bpidx, __builtin_bit_cast(int, t8[r]));
    tp[r] = __builtin_bit_cast(float, v);
  }
  v16bf b;
#pragma unroll
  for (int j = 0; j < 8; ++j){
    b[j]     = (__bf16)t8[j];
    b[j + 8] = (__bf16)tp[j];
  }
  return mask_frag(b, m);
}

__device__ __forceinline__ unsigned short bf16_bits(float f){
  __bf16 h = (__bf16)f;
  unsigned short u;
  __builtin_memcpy(&u, &h, 2);
  return u;
}

// ---------------- kernel 1: pack x into per-step WMMA-B fragment order ------
// xp[(t*TILES + tile)*FRAG + r] bf16, r<512: frag1 (c=0..31), r>=512: frag2
// (c=32..41 padded to 48, lanes 0..15 only).
__global__ __launch_bounds__(256) void pack_x(const float* __restrict__ x,
                                              unsigned int* __restrict__ xp32){
  __shared__ unsigned short buf[TT][Cn][16];   // [t][c][n] bf16 bits (43 KB)
  const int tid  = threadIdx.x;
  const int tile = blockIdx.x;
  const int t0   = blockIdx.y * TT;

  // phase 1: coalesced float4 reads along t, convert, scatter to LDS
  {
    const int tl = tid & 7;                    // 8 threads * float4 = 32 t's
    for (int pr = (tid >> 3); pr < 16 * Cn; pr += 32){
      const int n = pr & 15, c = pr >> 4;
      const float* src = x + ((size_t)(tile*16 + n) * Cn + c) * Tn + t0 + tl*4;
      float v[4] = {0.f, 0.f, 0.f, 0.f};
      if (t0 + tl*4 + 3 < Tn){
        float4 q = *(const float4*)src;
        v[0] = q.x; v[1] = q.y; v[2] = q.z; v[3] = q.w;
      } else {
#pragma unroll
        for (int i = 0; i < 4; ++i) if (t0 + tl*4 + i < Tn) v[i] = src[i];
      }
#pragma unroll
      for (int i = 0; i < 4; ++i) buf[tl*4 + i][c][n] = bf16_bits(v[i]);
    }
  }
  __syncthreads();

  // phase 2: emit fragments, coalesced 4B stores
  for (int idx = tid; idx < TT * (FRAG/2); idx += 256){
    const int t = idx / (FRAG/2);
    if (t0 + t >= Tn) continue;
    const int pr = idx % (FRAG/2);
    unsigned int w = 0;
#pragma unroll
    for (int k = 0; k < 2; ++k){
      const int r = pr*2 + k;
      unsigned int bits;
      if (r < 512){                              // frag1: c = (L>>4)*16 + j
        const int L = r >> 4, j = r & 15;
        const int c = ((L >> 4) << 4) + j, n = L & 15;
        bits = buf[t][c][n];
      } else {                                   // frag2: lanes 0..15, c=32+j
        const int q = r - 512;
        const int j = q & 15, n = (q >> 4) & 15;
        const int c = 32 + j;
        bits = (c < Cn) ? (unsigned int)buf[t][c][n] : 0u;
      }
      w |= bits << (16 * k);
    }
    xp32[((size_t)(t0 + t) * TILES + tile) * (FRAG/2) + pr] = w;
  }
}

// ---------------- kernel 2: the recurrent scan (one wave per batch tile) ----
template<bool PACKED>
__global__ __launch_bounds__(32) void rnn_scan(
    const float* __restrict__ x,
    const float* __restrict__ h0,
    const float* __restrict__ Wih0, const float* __restrict__ Whh0,
    const float* __restrict__ bih0, const float* __restrict__ bhh0,
    const float* __restrict__ Wih1, const float* __restrict__ Whh1,
    const float* __restrict__ bih1, const float* __restrict__ bhh1,
    const float* __restrict__ Wfc,  const float* __restrict__ bfc,
    const __bf16* __restrict__ xp,
    float* __restrict__ out)
{
  const int lane = threadIdx.x & 31;
  const int tile = blockIdx.x;
  const int n    = lane & 15;
  const int hi   = lane >> 4;
  const unsigned m     = (hi == 0) ? 0xFFFFFFFFu : 0u;  // B-frag live mask
  const int      bpidx = ((lane & 15) + 16) * 4;        // bpermute byte index

  // Constant A operands (M=hidden, K padded to 32), built once per wave.
  // A layout: lane holds row M=lane&15; element j has K=j+16*(lane>>4)
  v16bf Aih0a, Aih0b, Ahh0, Aih1, Ahh1;
#pragma unroll
  for (int j = 0; j < 16; ++j){
    const int k  = j + hi*16;   // 0..31
    Aih0a[j] = (__bf16)Wih0[n*Cn + k];
    const int c2 = 32 + k;
    Aih0b[j] = (c2 < Cn) ? (__bf16)Wih0[n*Cn + c2] : (__bf16)0.0f;
    Ahh0[j]  = (k < Hn) ? (__bf16)Whh0[n*Hn + k] : (__bf16)0.0f;
    Aih1[j]  = (k < Hn) ? (__bf16)Wih1[n*Hn + k] : (__bf16)0.0f;
    Ahh1[j]  = (k < Hn) ? (__bf16)Whh1[n*Hn + k] : (__bf16)0.0f;
  }

  // Bias in C/D layout: element r is hidden index r + 8*hi, same for all N.
  v8f bias0, bias1;
#pragma unroll
  for (int r = 0; r < 8; ++r){
    const int h = r + hi*8;
    bias0[r] = bih0[h] + bhh0[h];
    bias1[r] = bih1[h] + bhh1[h];
  }

  // Recurrent state as B operands (transposed: K=hidden, N=batch).
  v16bf g0, g1;
#pragma unroll
  for (int j = 0; j < 16; ++j){
    g0[j] = (__bf16)h0[0*Bn*Hn + (tile*16 + n)*Hn + (j & 15)];
    g1[j] = (__bf16)h0[1*Bn*Hn + (tile*16 + n)*Hn + (j & 15)];
  }
  g0 = mask_frag(g0, m);
  g1 = mask_frag(g1, m);

  // Branch-free x-fragment load (upper half-wave reads partner bytes, masked).
  auto load_bx = [&](int t, v16bf& b1, v16bf& b2){
    if constexpr (PACKED){
      const __bf16* base = xp + ((size_t)t * TILES + tile) * FRAG;
      b1 = *(const v16bf*)(base + lane*16);               // 32B/lane, coalesced
      b2 = mask_frag(*(const v16bf*)(base + 512 + (lane & 15)*16), m);
    } else {
      const float* xb = x + (size_t)(tile*16 + n) * Cn * Tn + t;
#pragma unroll
      for (int j = 0; j < 16; ++j) b1[j] = (__bf16)xb[(hi*16 + j) * Tn];
#pragma unroll
      for (int j = 0; j < 16; ++j){
        const int c = 32 + j;
        b2[j] = (c < Cn) ? (__bf16)xb[c * Tn] : (__bf16)0.0f;
      }
      b2 = mask_frag(b2, m);
    }
  };

  // Software pipeline: ax(t) = bias0 + W_ih0 · x_tT is independent of the
  // recurrence, so it is computed one step ahead and overlaps the serial
  // wmma->tanh->bpermute chain.
  v16bf cx1, cx2;
  load_bx(0, cx1, cx2);
  v8f ax = bias0;
  ax = wmma_bf16(Aih0a, cx1, ax);
  ax = wmma_bf16(Aih0b, cx2, ax);

  float t1[8];
  for (int t = 0; t < Tn; ++t){
    const int tnext = (t + 1 < Tn) ? t + 1 : t;
    load_bx(tnext, cx1, cx2);                   // prefetch next step's x

    // layer 0 recurrent part (on the chain) + layer 1 recurrent part (free).
    v8f a0 = wmma_bf16(Ahh0, g0, ax);
    v8f a1 = wmma_bf16(Ahh1, g1, bias1);

    float t0v[8];
#pragma unroll
    for (int r = 0; r < 8; ++r) t0v[r] = tanh_f32(a0[r]);

    // next step's x contribution: independent, fills WMMA->VALU hazard slots.
    v8f axn = bias0;
    axn = wmma_bf16(Aih0a, cx1, axn);
    axn = wmma_bf16(Aih0b, cx2, axn);

    g0 = d_to_bfrag(t0v, bpidx, m);
    a1 = wmma_bf16(Aih1, g0, a1);
#pragma unroll
    for (int r = 0; r < 8; ++r) t1[r] = tanh_f32(a1[r]);
    g1 = d_to_bfrag(t1, bpidx, m);

    ax = axn;
  }

  // FC epilogue: lanes 0..15 each own batch column n (f32 path, no bf16 loss).
  float hc[16];
#pragma unroll
  for (int r = 0; r < 8; ++r){
    int v = __builtin_amdgcn_ds_bpermute(bpidx, __builtin_bit_cast(int, t1[r]));
    hc[8 + r] = __builtin_bit_cast(float, v);
    hc[r]     = t1[r];
  }
  if (hi == 0){
    const int b = tile*16 + n;
    float o0 = bfc[0], o1 = bfc[1];
#pragma unroll
    for (int j = 0; j < 16; ++j){
      o0 += Wfc[j]      * hc[j];
      o1 += Wfc[Hn + j] * hc[j];
    }
    out[b*NCn + 0] = o0;
    out[b*NCn + 1] = o1;
  }
}

// ---------------- launch ----------------
extern "C" void kernel_launch(void* const* d_in, const int* in_sizes, int n_in,
                              void* d_out, int out_size, void* d_ws, size_t ws_size,
                              hipStream_t stream){
  const float* x    = (const float*)d_in[0];
  const float* h0   = (const float*)d_in[1];
  const float* Wih0 = (const float*)d_in[2];
  const float* Whh0 = (const float*)d_in[3];
  const float* bih0 = (const float*)d_in[4];
  const float* bhh0 = (const float*)d_in[5];
  const float* Wih1 = (const float*)d_in[6];
  const float* Whh1 = (const float*)d_in[7];
  const float* bih1 = (const float*)d_in[8];
  const float* bhh1 = (const float*)d_in[9];
  const float* Wfc  = (const float*)d_in[10];
  const float* bfc  = (const float*)d_in[11];
  float* out = (float*)d_out;
  (void)in_sizes; (void)n_in; (void)out_size;

  const size_t need = (size_t)Tn * TILES * FRAG * 2;   // ~157 MB packed bf16
  if (ws_size >= need){
    dim3 g(TILES, (Tn + TT - 1) / TT);
    pack_x<<<g, 256, 0, stream>>>(x, (unsigned int*)d_ws);
    rnn_scan<true><<<TILES, 32, 0, stream>>>(x, h0, Wih0, Whh0, bih0, bhh0,
                                             Wih1, Whh1, bih1, bhh1, Wfc, bfc,
                                             (const __bf16*)d_ws, out);
  } else {
    rnn_scan<false><<<TILES, 32, 0, stream>>>(x, h0, Wih0, Whh0, bih0, bhh0,
                                              Wih1, Whh1, bih1, bhh1, Wfc, bfc,
                                              nullptr, out);
  }
}
